// Gemma3Model_29832842838493
// MI455X (gfx1250) — compile-verified, matrix operand order checked by reading, not run on hardware
//
#include <hip/hip_runtime.h>
#include <stdint.h>

// ---------------------------------------------------------------------------
// Types
// ---------------------------------------------------------------------------
typedef __attribute__((ext_vector_type(16))) __bf16 v16bf;
typedef __attribute__((ext_vector_type(8)))  float  v8f;
typedef __attribute__((ext_vector_type(4)))  unsigned int u32x4;
typedef __attribute__((ext_vector_type(4)))  int    i32x4;
typedef __attribute__((ext_vector_type(4)))  float  f32x4;

#define AS1 __attribute__((address_space(1)))
#define AS3 __attribute__((address_space(3)))

#define S_LEN   2048
#define EMB_D   1024
#define N_HEADS 8
#define N_KV    4
#define HEAD_D  128
#define HIDDEN  4096
#define VOCAB   32000

// CDNA5 async global->LDS (DMA) path, if the toolchain exposes it.
#if defined(__has_builtin)
# if __has_builtin(__builtin_amdgcn_global_load_async_to_lds_b128)
#  define HAVE_ASYNC_LDS 1
# endif
#endif
#ifndef HAVE_ASYNC_LDS
# define HAVE_ASYNC_LDS 0
#endif

// ---------------------------------------------------------------------------
// Generic bf16 WMMA GEMM:  C(f32, MxN) = A(bf16, MxK, row-major, lda)
//                                      * B(bf16, KxN, row-major, ldb)
// Requires: M % 128 == 0, N % 128 == 0, K % 32 == 0 (true for every GEMM here)
// Block: 256 threads = 8 waves; block tile 128x128, K-step 32.
// Double-buffered LDS pipeline: issue next tile's loads before the WMMA
// block, commit after it, so global latency hides under matrix math.
// A tile goes global->LDS via async DMA when available (ASYNCcnt), B tile is
// register-staged and transposed into LDS so bf16 K-pairs are contiguous.
// ---------------------------------------------------------------------------
#define TM 128
#define TN 128
#define TK 32
#define LDSP 40   // padded LDS row stride (bf16 elements)

__global__ __launch_bounds__(256)
void gemm_bf16_wmma(const __bf16* __restrict__ A, int lda,
                    const __bf16* __restrict__ B, int ldb,
                    float* __restrict__ C, int ldc,
                    int M, int N, int K)
{
    (void)M;
    __shared__ __bf16 As[2][TM * LDSP];   // [buf][row][k]
    __shared__ __bf16 Bs[2][TN * LDSP];   // [buf][n][k]  (transposed)

    const int tid  = threadIdx.x;
    const int lane = tid & 31;
    const int w    = tid >> 5;     // 0..7
    const int wr   = w & 1;        // wave row group (64 rows each)
    const int wc   = w >> 1;       // wave col group (32 cols each)
    const int lm   = lane & 15;
    const int lh   = lane >> 4;    // lane half
    const int m0   = blockIdx.y * TM;
    const int n0   = blockIdx.x * TN;

    v8f acc[4][2] = {};

    // staging registers (A fallback path + B path)
    u32x4 aR0 = {}, aR1 = {}, bR0 = {}, bR1 = {};

    auto issueA = [&](int buf, int kb) {
#pragma unroll
        for (int i = 0; i < 2; ++i) {
            int c   = tid + (i << 8);
            int row = c >> 2;
            int kc  = (c & 3) << 3;
            const __bf16* src = A + (size_t)(m0 + row) * lda + kb + kc;
#if HAVE_ASYNC_LDS
            __builtin_amdgcn_global_load_async_to_lds_b128(
                (AS1 i32x4*)src,
                (AS3 i32x4*)&As[buf][row * LDSP + kc],
                0, 0);
#else
            if (i) aR1 = *(const u32x4*)src; else aR0 = *(const u32x4*)src;
#endif
        }
#if HAVE_ASYNC_LDS
        (void)buf;
#endif
    };
    auto commitA = [&](int buf) {
#if !HAVE_ASYNC_LDS
#pragma unroll
        for (int i = 0; i < 2; ++i) {
            int c   = tid + (i << 8);
            int row = c >> 2;
            int kc  = (c & 3) << 3;
            *(u32x4*)&As[buf][row * LDSP + kc] = i ? aR1 : aR0;
        }
#else
        (void)buf;
#endif
    };
    auto issueB = [&](int kb) {
#pragma unroll
        for (int i = 0; i < 2; ++i) {
            int c    = tid + (i << 8);
            int krow = c >> 4;
            int nc   = (c & 15) << 3;
            const __bf16* src = B + (size_t)(kb + krow) * ldb + n0 + nc;
            if (i) bR1 = *(const u32x4*)src; else bR0 = *(const u32x4*)src;
        }
    };
    auto commitB = [&](int buf) {
#pragma unroll
        for (int i = 0; i < 2; ++i) {
            int c    = tid + (i << 8);
            int krow = c >> 4;
            int nc   = (c & 15) << 3;
            union { u32x4 q; __bf16 h[8]; } u;
            u.q = i ? bR1 : bR0;
#pragma unroll
            for (int j = 0; j < 8; ++j)
                Bs[buf][(nc + j) * LDSP + krow] = u.h[j];
        }
    };
    auto waitAsync = [&]() {
#if HAVE_ASYNC_LDS
        asm volatile("s_wait_asynccnt 0" ::: "memory");
#endif
    };

    const int nk = K / TK;

    // ---- prologue: stage tile 0 into buffer 0 ----
    issueA(0, 0);
    issueB(0);
    commitA(0);
    commitB(0);
    waitAsync();
    __syncthreads();

    for (int t = 0; t < nk; ++t) {
        const int  cur     = t & 1;
        const int  nxt     = cur ^ 1;
        const bool hasNext = (t + 1) < nk;

        // issue next tile's global traffic before the math
        if (hasNext) {
            issueA(nxt, (t + 1) * TK);
            issueB((t + 1) * TK);
        }

        // ---- fragments from current buffer ----
        union Frag { v16bf v; uint32_t u[8]; };
        Frag fa[4], fb[2];
#pragma unroll
        for (int mt = 0; mt < 4; ++mt) {
            const uint32_t* rp =
                (const uint32_t*)&As[cur][(wr * 64 + mt * 16 + lm) * LDSP];
#pragma unroll
            for (int v = 0; v < 8; ++v) {
                int k0 = ((v >> 2) << 4) + (lh << 3) + ((v & 3) << 1);
                fa[mt].u[v] = rp[k0 >> 1];
            }
        }
#pragma unroll
        for (int nt = 0; nt < 2; ++nt) {
            const uint32_t* rp =
                (const uint32_t*)&Bs[cur][(wc * 32 + nt * 16 + lm) * LDSP];
#pragma unroll
            for (int v = 0; v < 8; ++v) {
                int k0 = ((v >> 2) << 4) + (lh << 3) + ((v & 3) << 1);
                fb[nt].u[v] = rp[k0 >> 1];
            }
        }
        // ---- 8 WMMAs ----
#pragma unroll
        for (int mt = 0; mt < 4; ++mt)
#pragma unroll
            for (int nt = 0; nt < 2; ++nt)
                acc[mt][nt] = __builtin_amdgcn_wmma_f32_16x16x32_bf16(
                    false, fa[mt].v, false, fb[nt].v,
                    (short)0, acc[mt][nt], false, false);

        // commit next tile into the other buffer (loadcnt wait lands here,
        // after the WMMA block) and synchronize
        if (hasNext) {
            commitA(nxt);
            commitB(nxt);
            waitAsync();
            __syncthreads();
        }
    }

    // ---- store C (fp32). C layout: VGPR r -> M = r (+8 for hi lanes) ----
#pragma unroll
    for (int mt = 0; mt < 4; ++mt)
#pragma unroll
        for (int nt = 0; nt < 2; ++nt) {
            int col = n0 + wc * 32 + nt * 16 + lm;
#pragma unroll
            for (int r = 0; r < 8; ++r) {
                int rowm = m0 + wr * 64 + mt * 16 + lh * 8 + r;
                C[(size_t)rowm * ldc + col] = acc[mt][nt][r];
            }
        }
}

// ---------------------------------------------------------------------------
// Elementwise / fused kernels
// ---------------------------------------------------------------------------
__global__ void embed_kernel(const int* __restrict__ ids,
                             const float* __restrict__ emb,
                             float* __restrict__ x)
{
    int row = blockIdx.x;
    int id  = ids[row];
    const float* src = emb + (size_t)id * EMB_D;
    float* dst = x + (size_t)row * EMB_D;
    for (int j = threadIdx.x; j < EMB_D; j += 256)
        dst[j] = src[j] * 32.0f;   // sqrt(1024)
}

__global__ void f32_to_bf16_kernel(const float* __restrict__ in,
                                   __bf16* __restrict__ out, long n)
{
    long i = ((long)blockIdx.x * 256 + threadIdx.x) * 4;
    if (i + 3 < n) {
        f32x4 v = *(const f32x4*)(in + i);
        out[i + 0] = (__bf16)v.x;
        out[i + 1] = (__bf16)v.y;
        out[i + 2] = (__bf16)v.z;
        out[i + 3] = (__bf16)v.w;
    } else {
        for (; i < n; ++i) out[i] = (__bf16)in[i];
    }
}

__global__ void rmsnorm_bf16_kernel(const float* __restrict__ in,
                                    const float* __restrict__ g,
                                    __bf16* __restrict__ out, int D)
{
    __shared__ float red[256];
    int row = blockIdx.x;
    const float* p = in + (size_t)row * D;
    float s = 0.f;
    for (int j = threadIdx.x; j < D; j += 256) { float v = p[j]; s += v * v; }
    red[threadIdx.x] = s;
    __syncthreads();
    for (int st = 128; st > 0; st >>= 1) {
        if (threadIdx.x < st) red[threadIdx.x] += red[threadIdx.x + st];
        __syncthreads();
    }
    float rs = rsqrtf(red[0] / (float)D + 1e-6f);
    for (int j = threadIdx.x; j < D; j += 256)
        out[(size_t)row * D + j] = (__bf16)(p[j] * rs * (1.0f + g[j]));
}

// x += rmsnorm(a, g)
__global__ void rmsnorm_add_kernel(const float* __restrict__ a,
                                   const float* __restrict__ g,
                                   float* __restrict__ x, int D)
{
    __shared__ float red[256];
    int row = blockIdx.x;
    const float* p = a + (size_t)row * D;
    float s = 0.f;
    for (int j = threadIdx.x; j < D; j += 256) { float v = p[j]; s += v * v; }
    red[threadIdx.x] = s;
    __syncthreads();
    for (int st = 128; st > 0; st >>= 1) {
        if (threadIdx.x < st) red[threadIdx.x] += red[threadIdx.x + st];
        __syncthreads();
    }
    float rs = rsqrtf(red[0] / (float)D + 1e-6f);
    for (int j = threadIdx.x; j < D; j += 256)
        x[(size_t)row * D + j] += p[j] * rs * (1.0f + g[j]);
}

// Per (pos, head): rmsnorm(head_dim=128) with (1+g), RoPE, optional 1/16 scale.
// transposed==0: out[pos][h*128+d] (row-major, stride heads*128)
// transposed==1: out[(h*128+d)*S + pos]  (k stored transposed for scores GEMM)
__global__ void qk_rope_kernel(const float* __restrict__ in, int heads,
                               const float* __restrict__ g,
                               __bf16* __restrict__ out, int transposed,
                               float theta, float oscale)
{
    __shared__ float red[128];
    __shared__ float sh[128];
    int pos = blockIdx.x / heads;
    int h   = blockIdx.x % heads;
    int d   = threadIdx.x;           // 0..127
    int stride = heads * HEAD_D;
    float v = in[(size_t)pos * stride + h * HEAD_D + d];
    red[d] = v * v;
    __syncthreads();
    for (int st = 64; st > 0; st >>= 1) {
        if (d < st) red[d] += red[d + st];
        __syncthreads();
    }
    float rs = rsqrtf(red[0] / (float)HEAD_D + 1e-6f);
    sh[d] = v * rs * (1.0f + g[d]);
    __syncthreads();
    float rot = (d < 64) ? -sh[d + 64] : sh[d - 64];
    float ex  = (float)(2 * (d & 63)) / (float)HEAD_D;
    float inv = __expf(-ex * __logf(theta));
    float ang = (float)pos * inv;
    float o   = (sh[d] * cosf(ang) + rot * sinf(ang)) * oscale;
    if (transposed)
        out[((size_t)h * HEAD_D + d) * S_LEN + pos] = (__bf16)o;
    else
        out[(size_t)pos * stride + h * HEAD_D + d] = (__bf16)o;
}

// Masked softmax over one row of scores (len S). window>0 => sliding window.
__global__ void softmax_mask_kernel(const float* __restrict__ scores,
                                    __bf16* __restrict__ probs, int window)
{
    __shared__ float red[256];
    int i = blockIdx.x;
    const float* row = scores + (size_t)i * S_LEN;
    __bf16* orow = probs + (size_t)i * S_LEN;
    int lo = (window > 0) ? max(0, i - window + 1) : 0;

    float m = -1e30f;
    for (int j = lo + threadIdx.x; j <= i; j += 256) m = fmaxf(m, row[j]);
    red[threadIdx.x] = m;
    __syncthreads();
    for (int st = 128; st > 0; st >>= 1) {
        if (threadIdx.x < st)
            red[threadIdx.x] = fmaxf(red[threadIdx.x], red[threadIdx.x + st]);
        __syncthreads();
    }
    float rmax = red[0];
    __syncthreads();

    float s = 0.f;
    for (int j = lo + threadIdx.x; j <= i; j += 256) s += __expf(row[j] - rmax);
    red[threadIdx.x] = s;
    __syncthreads();
    for (int st = 128; st > 0; st >>= 1) {
        if (threadIdx.x < st) red[threadIdx.x] += red[threadIdx.x + st];
        __syncthreads();
    }
    float invs = 1.0f / red[0];

    for (int j = threadIdx.x; j < S_LEN; j += 256) {
        float p = (j >= lo && j <= i) ? __expf(row[j] - rmax) * invs : 0.f;
        orow[j] = (__bf16)p;
    }
}

// out = bf16( gelu_tanh(g1) * g2 )
__global__ void gelu_mul_kernel(const float* __restrict__ g1,
                                const float* __restrict__ g2,
                                __bf16* __restrict__ out, long n)
{
    long i = (long)blockIdx.x * 256 + threadIdx.x;
    if (i >= n) return;
    float x = g1[i];
    float t = 0.7978845608028654f * (x + 0.044715f * x * x * x);
    float gl = 0.5f * x * (1.0f + tanhf(t));
    out[i] = (__bf16)(gl * g2[i]);
}

// ---------------------------------------------------------------------------
// Host orchestration
// ---------------------------------------------------------------------------
static inline void gemm(const __bf16* A, int lda, const __bf16* B, int ldb,
                        float* C, int ldc, int M, int N, int K,
                        hipStream_t stream)
{
    dim3 grid(N / TN, M / TM);
    gemm_bf16_wmma<<<grid, 256, 0, stream>>>(A, lda, B, ldb, C, ldc, M, N, K);
}

extern "C" void kernel_launch(void* const* d_in, const int* in_sizes, int n_in,
                              void* d_out, int out_size, void* d_ws, size_t ws_size,
                              hipStream_t stream)
{
    (void)in_sizes; (void)n_in; (void)out_size; (void)ws_size;
    const int S = S_LEN;

    const int*   ids = (const int*)d_in[0];
    const float* tok = (const float*)d_in[1];
    // block params, per layer L (13 entries each):
    // 0 Wq, 1 Wk, 2 Wv, 3 Wo, 4 q_norm, 5 k_norm, 6 W1, 7 W2, 8 W3,
    // 9 input_ln, 10 post_attn_ln, 11 pre_ff_ln, 12 post_ff_ln
    auto P = [&](int L, int j) { return (const float*)d_in[2 + L * 13 + j]; };
    const float* final_norm = (const float*)d_in[2 + 4 * 13];
    const float* out_head_w = (const float*)d_in[2 + 4 * 13 + 1];
    float* out = (float*)d_out;

    // workspace bump allocator
    char* ws = (char*)d_ws;
    size_t off = 0;
    auto alloc = [&](size_t bytes) -> void* {
        void* p = ws + off;
        off += (bytes + 255) & ~(size_t)255;
        return p;
    };

    float*  x     = (float*)alloc((size_t)S * EMB_D * 4);
    __bf16* hb    = (__bf16*)alloc((size_t)S * EMB_D * 2);
    float*  qf    = (float*)alloc((size_t)S * EMB_D * 4);
    float*  kf    = (float*)alloc((size_t)S * N_KV * HEAD_D * 4);
    float*  vf    = (float*)alloc((size_t)S * N_KV * HEAD_D * 4);
    __bf16* qb    = (__bf16*)alloc((size_t)S * EMB_D * 2);
    __bf16* kT    = (__bf16*)alloc((size_t)N_KV * HEAD_D * S * 2);
    __bf16* vb    = (__bf16*)alloc((size_t)S * N_KV * HEAD_D * 2);
    float*  sc    = (float*)alloc((size_t)S * S * 4);
    __bf16* pb    = (__bf16*)alloc((size_t)S * S * 2);
    float*  ctx   = (float*)alloc((size_t)S * EMB_D * 4);
    __bf16* ctxb  = (__bf16*)alloc((size_t)S * EMB_D * 2);
    float*  af    = (float*)alloc((size_t)S * EMB_D * 4);
    float*  g1    = (float*)alloc((size_t)S * HIDDEN * 4);
    float*  g2    = (float*)alloc((size_t)S * HIDDEN * 4);
    __bf16* fbuf  = (__bf16*)alloc((size_t)S * HIDDEN * 2);

    __bf16 *wqb[4], *wkb[4], *wvb[4], *wob[4], *w1b[4], *w2b[4], *w3b[4];
    for (int L = 0; L < 4; ++L) {
        wqb[L] = (__bf16*)alloc((size_t)EMB_D * EMB_D * 2);
        wkb[L] = (__bf16*)alloc((size_t)EMB_D * N_KV * HEAD_D * 2);
        wvb[L] = (__bf16*)alloc((size_t)EMB_D * N_KV * HEAD_D * 2);
        wob[L] = (__bf16*)alloc((size_t)EMB_D * EMB_D * 2);
        w1b[L] = (__bf16*)alloc((size_t)EMB_D * HIDDEN * 2);
        w2b[L] = (__bf16*)alloc((size_t)EMB_D * HIDDEN * 2);
        w3b[L] = (__bf16*)alloc((size_t)HIDDEN * EMB_D * 2);
    }
    __bf16* ohb = (__bf16*)alloc((size_t)EMB_D * VOCAB * 2);

    auto conv = [&](const float* src, __bf16* dst, long n) {
        long blocks = (n / 4 + 255) / 256;
        f32_to_bf16_kernel<<<(int)blocks, 256, 0, stream>>>(src, dst, n);
    };

    // ---- weight conversion (fp32 -> bf16) ----
    for (int L = 0; L < 4; ++L) {
        conv(P(L, 0), wqb[L], (long)EMB_D * EMB_D);
        conv(P(L, 1), wkb[L], (long)EMB_D * N_KV * HEAD_D);
        conv(P(L, 2), wvb[L], (long)EMB_D * N_KV * HEAD_D);
        conv(P(L, 3), wob[L], (long)EMB_D * EMB_D);
        conv(P(L, 6), w1b[L], (long)EMB_D * HIDDEN);
        conv(P(L, 7), w2b[L], (long)EMB_D * HIDDEN);
        conv(P(L, 8), w3b[L], (long)HIDDEN * EMB_D);
    }
    conv(out_head_w, ohb, (long)EMB_D * VOCAB);

    // ---- embedding ----
    embed_kernel<<<S, 256, 0, stream>>>(ids, tok, x);

    // ---- transformer blocks ----
    for (int L = 0; L < 4; ++L) {
        bool  local = (L != 3);                 // layers 0-2 sliding, 3 full
        float theta = local ? 10000.0f : 1000000.0f;

        rmsnorm_bf16_kernel<<<S, 256, 0, stream>>>(x, P(L, 9), hb, EMB_D);

        gemm(hb, EMB_D, wqb[L], EMB_D, qf, EMB_D, S, EMB_D, EMB_D, stream);
        gemm(hb, EMB_D, wkb[L], N_KV * HEAD_D, kf, N_KV * HEAD_D,
             S, N_KV * HEAD_D, EMB_D, stream);
        gemm(hb, EMB_D, wvb[L], N_KV * HEAD_D, vf, N_KV * HEAD_D,
             S, N_KV * HEAD_D, EMB_D, stream);

        qk_rope_kernel<<<S * N_HEADS, 128, 0, stream>>>(
            qf, N_HEADS, P(L, 4), qb, 0, theta, 0.0625f); // * 256^-0.5
        qk_rope_kernel<<<S * N_KV, 128, 0, stream>>>(
            kf, N_KV, P(L, 5), kT, 1, theta, 1.0f);
        conv(vf, vb, (long)S * N_KV * HEAD_D);

        for (int h = 0; h < N_HEADS; ++h) {
            int kvh = h >> 1;   // GROUP = 2
            // scores = q_h (S x 128) * kT_kvh (128 x S)
            gemm(qb + h * HEAD_D, EMB_D,
                 kT + (size_t)kvh * HEAD_D * S, S,
                 sc, S, S, S, HEAD_D, stream);
            softmax_mask_kernel<<<S, 256, 0, stream>>>(sc, pb, local ? 512 : 0);
            // ctx_h = P (S x S) * v_kvh (S x 128)
            gemm(pb, S, vb + kvh * HEAD_D, N_KV * HEAD_D,
                 ctx + h * HEAD_D, EMB_D, S, HEAD_D, S, stream);
        }

        conv(ctx, ctxb, (long)S * EMB_D);
        gemm(ctxb, EMB_D, wob[L], EMB_D, af, EMB_D, S, EMB_D, EMB_D, stream);
        rmsnorm_add_kernel<<<S, 256, 0, stream>>>(af, P(L, 10), x, EMB_D);

        rmsnorm_bf16_kernel<<<S, 256, 0, stream>>>(x, P(L, 11), hb, EMB_D);
        gemm(hb, EMB_D, w1b[L], HIDDEN, g1, HIDDEN, S, HIDDEN, EMB_D, stream);
        gemm(hb, EMB_D, w2b[L], HIDDEN, g2, HIDDEN, S, HIDDEN, EMB_D, stream);
        {
            long n = (long)S * HIDDEN;
            gelu_mul_kernel<<<(int)((n + 255) / 256), 256, 0, stream>>>(
                g1, g2, fbuf, n);
        }
        gemm(fbuf, HIDDEN, w3b[L], EMB_D, af, EMB_D, S, EMB_D, HIDDEN, stream);
        rmsnorm_add_kernel<<<S, 256, 0, stream>>>(af, P(L, 12), x, EMB_D);
    }

    // ---- final norm + LM head ----
    rmsnorm_bf16_kernel<<<S, 256, 0, stream>>>(x, final_norm, hb, EMB_D);
    gemm(hb, EMB_D, ohb, VOCAB, out, VOCAB, S, VOCAB, EMB_D, stream);
}
